// Copy6ExtendedCQCNN_61375082660336
// MI455X (gfx1250) — compile-verified, matrix operand order checked by reading, not run on hardware
//
#include <hip/hip_runtime.h>
#include <cstddef>
#include <cstdint>

typedef __bf16 bf16_t;
typedef __attribute__((ext_vector_type(16))) __bf16 v16bf;
typedef __attribute__((ext_vector_type(8)))  float  v8f;
typedef int v4i_vs __attribute__((vector_size(16)));  // matches builtin param type

#if defined(__has_builtin)
#if __has_builtin(__builtin_amdgcn_global_load_async_to_lds_b128)
#define ASYNC_LDS 1
#endif
#endif
#ifndef ASYNC_LDS
#define ASYNC_LDS 0
#endif

#define AS1 __attribute__((address_space(1)))
#define AS3 __attribute__((address_space(3)))

__device__ __forceinline__ v8f zero8() {
  v8f z;
#pragma unroll
  for (int i = 0; i < 8; ++i) z[i] = 0.0f;
  return z;
}

// 16-byte global -> LDS copy. Async (ASYNCcnt, no VGPR data) when available.
__device__ __forceinline__ void copy16_g2l(const bf16_t* g, bf16_t* l) {
#if ASYNC_LDS
  // AS-specific pointers implicitly convert to generic (superset) on AMDGPU,
  // so these casts are valid whether the builtin's params are generic or
  // AS1/AS3-qualified int4 pointers.
  __builtin_amdgcn_global_load_async_to_lds_b128((AS1 v4i_vs*)g,
                                                 (AS3 v4i_vs*)l, 0, 0);
#else
  *(uint4*)l = *(const uint4*)g;
#endif
}

__device__ __forceinline__ void wait_async_all() {
#if ASYNC_LDS
#if __has_builtin(__builtin_amdgcn_s_wait_asynccnt)
  __builtin_amdgcn_s_wait_asynccnt(0);
#else
  asm volatile("s_wait_asynccnt 0" ::: "memory");
#endif
#endif
}

// CDNA5 16-bit operand fragment (16x32 A or 32x16 B stored [row][k] in LDS):
// lanes 0-15: row = lane, K 0..7 in v0..v3 (pairs), K 16..23 in v4..v7
// lanes 16-31: same rows, K offset +8 (8..15 / 24..31)
__device__ __forceinline__ v16bf load_frag(const bf16_t* base, int row, int stride, int half) {
  v16bf f;
  const bf16_t* p = base + row * stride + half * 8;
#pragma unroll
  for (int i = 0; i < 4; ++i) {
    f[2 * i]     = p[2 * i];
    f[2 * i + 1] = p[2 * i + 1];
  }
#pragma unroll
  for (int i = 0; i < 4; ++i) {
    f[8 + 2 * i]     = p[16 + 2 * i];
    f[8 + 2 * i + 1] = p[16 + 2 * i + 1];
  }
  return f;
}

// ---------------------------------------------------------------------------
// GEMM1: A1 = relu( state[B,252] @ W1[128,252]^T + b1 ), f32 in, bf16 out,
// per-feature stats. Synchronous staging (needs f32->bf16 cvt + K padding).
// ---------------------------------------------------------------------------
__global__ __launch_bounds__(256) void gemm1_kernel(
    const float* __restrict__ X, const float* __restrict__ W,
    const float* __restrict__ bias, bf16_t* __restrict__ Out,
    float* __restrict__ gSum, float* __restrict__ gSq) {
  constexpr int N = 128, KSTEPS = 8, Kin = 252;
  __shared__ bf16_t Atile[128][40];
  __shared__ bf16_t Btile[N][40];
  __shared__ float sSum[N];
  __shared__ float sSq[N];

  const int tid  = threadIdx.x;
  const int lane = tid & 31;
  const int wave = tid >> 5;
  const int half = lane >> 4;
  const int m0   = blockIdx.x * 128;

  if (tid < N) { sSum[tid] = 0.0f; sSq[tid] = 0.0f; }

  v8f acc[N / 16];
#pragma unroll
  for (int t = 0; t < N / 16; ++t) acc[t] = zero8();

  for (int ks = 0; ks < KSTEPS; ++ks) {
    __syncthreads();
    const int kb = ks * 32;
    for (int idx = tid; idx < 128 * 32; idx += 256) {
      const int r = idx >> 5, k = idx & 31, kg = kb + k;
      float v = 0.0f;
      if (kg < Kin) {
        v = X[(size_t)(m0 + r) * Kin + kg];
        if (kg + 32 < Kin) __builtin_prefetch(&X[(size_t)(m0 + r) * Kin + kg + 32], 0, 0);
      }
      Atile[r][k] = (bf16_t)v;
    }
    for (int idx = tid; idx < N * 32; idx += 256) {
      const int n = idx >> 5, k = idx & 31, kg = kb + k;
      Btile[n][k] = (kg < Kin) ? (bf16_t)W[(size_t)n * Kin + kg] : (bf16_t)0.0f;
    }
    __syncthreads();
    const v16bf a = load_frag(&Atile[wave * 16][0], lane & 15, 40, half);
#pragma unroll
    for (int t = 0; t < N / 16; ++t) {
      const v16bf b = load_frag(&Btile[t * 16][0], lane & 15, 40, half);
      acc[t] = __builtin_amdgcn_wmma_f32_16x16x32_bf16(false, a, false, b,
                                                       (short)0, acc[t], false, false);
    }
  }
#pragma unroll
  for (int t = 0; t < N / 16; ++t) {
    const int col = t * 16 + (lane & 15);
    const float bv = bias[col];
    float ps = 0.0f, pq = 0.0f;
#pragma unroll
    for (int r = 0; r < 8; ++r) {
      float v = acc[t][r] + bv;
      v = v > 0.0f ? v : 0.0f;
      const int rowg = m0 + wave * 16 + half * 8 + r;
      Out[(size_t)rowg * N + col] = (bf16_t)v;
      ps += v;
      pq += v * v;
    }
    atomicAdd(&sSum[col], ps);
    atomicAdd(&sSq[col], pq);
  }
  __syncthreads();
  if (tid < N) {
    atomicAdd(&gSum[tid], sSum[tid]);
    atomicAdd(&gSq[tid], sSq[tid]);
  }
}

// ---------------------------------------------------------------------------
// Fold BN affine into weights: Wb[n,k] = bf16(W[n,k]*a[k]);
// biasF[n] = bias[n] + sum_k W[n,k]*s[k]. (tiny, one block)
// ---------------------------------------------------------------------------
template <int N, int K>
__global__ void fold_weight_kernel(const float* __restrict__ W,
                                   const float* __restrict__ bias,
                                   const float* __restrict__ aa,
                                   const float* __restrict__ ss,
                                   bf16_t* __restrict__ Wb,
                                   float* __restrict__ biasF) {
  const int n = blockIdx.x * blockDim.x + threadIdx.x;
  if (n >= N) return;
  float acc = bias[n];
  for (int k = 0; k < K; ++k) {
    const float w = W[n * K + k];
    Wb[n * K + k] = (bf16_t)(w * aa[k]);
    acc += w * ss[k];
  }
  biasF[n] = acc;
}

// ---------------------------------------------------------------------------
// Async double-buffered bf16 GEMM: Out = relu( X[B,K] @ Wb[N,K]^T + biasF ),
// all-bf16 tiles staged with global_load_async_to_lds_b128, + BN stats.
// ---------------------------------------------------------------------------
template <int N, int KSTEPS>
__global__ __launch_bounds__(256) void gemm_async_kernel(
    const bf16_t* __restrict__ X, const bf16_t* __restrict__ Wb,
    const float* __restrict__ biasF, bf16_t* __restrict__ Out,
    float* __restrict__ gSum, float* __restrict__ gSq) {
  constexpr int K = KSTEPS * 32;
  constexpr int NT = N / 16;
  __shared__ bf16_t Atile[2][128][40];
  __shared__ bf16_t Btile[2][N][40];
  __shared__ float sSum[N];
  __shared__ float sSq[N];

  const int tid  = threadIdx.x;
  const int lane = tid & 31;
  const int wave = tid >> 5;
  const int half = lane >> 4;
  const int m0   = blockIdx.x * 128;

  if (tid < N) { sSum[tid] = 0.0f; sSq[tid] = 0.0f; }

  auto stage = [&](int buf, int ks) {
    const int kb = ks * 32;
    for (int idx = tid; idx < 128 * 4; idx += 256) {       // A: 128 x 4 chunks
      const int r = idx >> 2, c = idx & 3;
      copy16_g2l(X + (size_t)(m0 + r) * K + kb + c * 8, &Atile[buf][r][c * 8]);
    }
    for (int idx = tid; idx < N * 4; idx += 256) {         // B: N x 4 chunks
      const int n = idx >> 2, c = idx & 3;
      copy16_g2l(Wb + (size_t)n * K + kb + c * 8, &Btile[buf][n][c * 8]);
    }
  };

  v8f acc[NT];
#pragma unroll
  for (int t = 0; t < NT; ++t) acc[t] = zero8();

  stage(0, 0);
  for (int ks = 0; ks < KSTEPS; ++ks) {
    const int cur = ks & 1;
    wait_async_all();      // my copies into buf[cur] have landed
    __syncthreads();       // everyone's copies landed; prior reads of buf[cur^1] done
    if (ks + 1 < KSTEPS) stage(cur ^ 1, ks + 1);  // DMA next tile while we compute
    const v16bf a = load_frag(&Atile[cur][wave * 16][0], lane & 15, 40, half);
#pragma unroll
    for (int t = 0; t < NT; ++t) {
      const v16bf b = load_frag(&Btile[cur][t * 16][0], lane & 15, 40, half);
      acc[t] = __builtin_amdgcn_wmma_f32_16x16x32_bf16(false, a, false, b,
                                                       (short)0, acc[t], false, false);
    }
  }
#pragma unroll
  for (int t = 0; t < NT; ++t) {
    const int col = t * 16 + (lane & 15);
    const float bv = biasF[col];
    float ps = 0.0f, pq = 0.0f;
#pragma unroll
    for (int r = 0; r < 8; ++r) {
      float v = acc[t][r] + bv;
      v = v > 0.0f ? v : 0.0f;
      const int rowg = m0 + wave * 16 + half * 8 + r;
      Out[(size_t)rowg * N + col] = (bf16_t)v;
      ps += v;
      pq += v * v;
    }
    atomicAdd(&sSum[col], ps);
    atomicAdd(&sSq[col], pq);
  }
  __syncthreads();
  if (tid < N) {
    atomicAdd(&gSum[tid], sSum[tid]);
    atomicAdd(&gSq[tid], sSq[tid]);
  }
}

// ---------------------------------------------------------------------------
// Fused middle: BN2-affine -> 64->32 tanh -> 32->4 tanh -> 4-qubit circuit ->
// <Z_w> -> V1 (4->64) relu -> bf16 store + BN3 stats. One sample per thread.
// ---------------------------------------------------------------------------
__global__ __launch_bounds__(256) void middle_kernel(
    const bf16_t* __restrict__ A2, const float* __restrict__ a2,
    const float* __restrict__ s2, const float* __restrict__ W3,
    const float* __restrict__ b3, const float* __restrict__ W4,
    const float* __restrict__ b4, const float* __restrict__ qp,
    const float* __restrict__ V1, const float* __restrict__ c1,
    bf16_t* __restrict__ Y1, float* __restrict__ gSum, float* __restrict__ gSq) {
  __shared__ float sW3[32 * 64];
  __shared__ float sb3[32];
  __shared__ float sW4[4 * 32];
  __shared__ float sb4[4];
  __shared__ float sV1[64 * 4];
  __shared__ float sc1[64];
  __shared__ float sA[64], sS[64];
  __shared__ float sQP[16];
  __shared__ float sSum[64], sSq[64];

  const int tid = threadIdx.x;
  for (int i = tid; i < 32 * 64; i += 256) sW3[i] = W3[i];
  for (int i = tid; i < 64 * 4; i += 256) sV1[i] = V1[i];
  if (tid < 128) sW4[tid] = W4[tid];
  if (tid < 32) sb3[tid] = b3[tid];
  if (tid < 4) sb4[tid] = b4[tid];
  if (tid < 64) {
    sc1[tid] = c1[tid];
    sA[tid] = a2[tid];
    sS[tid] = s2[tid];
    sSum[tid] = 0.0f;
    sSq[tid] = 0.0f;
  }
  if (tid < 16) sQP[tid] = qp[tid];
  __syncthreads();

  const size_t i = (size_t)blockIdx.x * 256 + tid;

  float x[64];
#pragma unroll
  for (int k = 0; k < 64; ++k) x[k] = sA[k] * (float)A2[i * 64 + k] + sS[k];

  float h[32];
#pragma unroll 4
  for (int j = 0; j < 32; ++j) {
    float acc = sb3[j];
#pragma unroll
    for (int k = 0; k < 64; ++k) acc += sW3[j * 64 + k] * x[k];
    h[j] = tanhf(acc);
  }
  float f4[4];
#pragma unroll
  for (int j = 0; j < 4; ++j) {
    float acc = sb4[j];
#pragma unroll
    for (int k = 0; k < 32; ++k) acc += sW4[j * 32 + k] * h[k];
    f4[j] = tanhf(acc);
  }

  // 4-qubit statevector, q0 is MSB (stride 8>>w), psi = |0000>
  float yr[16], yi[16];
#pragma unroll
  for (int k = 0; k < 16; ++k) { yr[k] = 0.0f; yi[k] = 0.0f; }
  yr[0] = 1.0f;

#pragma unroll
  for (int w = 0; w < 4; ++w) {  // data-encoding RY(f*pi)
    const float th = f4[w] * 3.14159265358979323846f * 0.5f;
    const float c = cosf(th), s = sinf(th);
    const int st = 8 >> w;
#pragma unroll
    for (int idx = 0; idx < 16; ++idx) {
      if ((idx & st) == 0) {
        const int j = idx | st;
        const float r0 = yr[idx], i0 = yi[idx], r1 = yr[j], i1 = yi[j];
        yr[idx] = c * r0 - s * r1;
        yi[idx] = c * i0 - s * i1;
        yr[j]   = s * r0 + c * r1;
        yi[j]   = s * i0 + c * i1;
      }
    }
  }
#pragma unroll
  for (int l = 0; l < 2; ++l) {  // variational RZ@RY + circular CNOTs
#pragma unroll
    for (int w = 0; w < 4; ++w) {
      const float ty = sQP[l * 8 + w * 2 + 0] * 0.5f;
      const float tz = sQP[l * 8 + w * 2 + 1] * 0.5f;
      const float c = cosf(ty), s = sinf(ty);
      const float cz = cosf(tz), sz = sinf(tz);
      const float g00r =  cz * c, g00i = -sz * c;
      const float g01r = -cz * s, g01i =  sz * s;
      const float g10r =  cz * s, g10i =  sz * s;
      const float g11r =  cz * c, g11i =  sz * c;
      const int st = 8 >> w;
#pragma unroll
      for (int idx = 0; idx < 16; ++idx) {
        if ((idx & st) == 0) {
          const int j = idx | st;
          const float r0 = yr[idx], i0 = yi[idx], r1 = yr[j], i1 = yi[j];
          yr[idx] = g00r * r0 - g00i * i0 + g01r * r1 - g01i * i1;
          yi[idx] = g00r * i0 + g00i * r0 + g01r * i1 + g01i * r1;
          yr[j]   = g10r * r0 - g10i * i0 + g11r * r1 - g11i * i1;
          yi[j]   = g10r * i0 + g10i * r0 + g11r * i1 + g11i * r1;
        }
      }
    }
#pragma unroll
    for (int w = 0; w < 4; ++w) {
      const int sc = 8 >> w;
      const int st = 8 >> ((w + 1) & 3);
#pragma unroll
      for (int idx = 0; idx < 16; ++idx) {
        if ((idx & sc) && !(idx & st)) {
          const int j = idx | st;
          const float tr = yr[idx], ti = yi[idx];
          yr[idx] = yr[j]; yi[idx] = yi[j];
          yr[j] = tr;      yi[j] = ti;
        }
      }
    }
  }
  float q[4];
#pragma unroll
  for (int w = 0; w < 4; ++w) {
    const int st = 8 >> w;
    float e = 0.0f;
#pragma unroll
    for (int idx = 0; idx < 16; ++idx) {
      const float p = yr[idx] * yr[idx] + yi[idx] * yi[idx];
      e += (idx & st) ? -p : p;
    }
    q[w] = e;
  }
  for (int j = 0; j < 64; ++j) {  // V1: 4 -> 64, relu, BN3 stats
    float acc = sc1[j];
#pragma unroll
    for (int k = 0; k < 4; ++k) acc += sV1[j * 4 + k] * q[k];
    acc = acc > 0.0f ? acc : 0.0f;
    Y1[i * 64 + j] = (bf16_t)acc;
    atomicAdd(&sSum[j], acc);
    atomicAdd(&sSq[j], acc * acc);
  }
  __syncthreads();
  if (tid < 64) {
    atomicAdd(&gSum[tid], sSum[tid]);
    atomicAdd(&gSq[tid], sSq[tid]);
  }
}

// ---------------------------------------------------------------------------
// Final: Y2 (raw bf16, BN folded into Vb3) -> V3 WMMA relu -> V4 (64->36) out.
// Async double-buffered staging.
// ---------------------------------------------------------------------------
__global__ __launch_bounds__(256) void final_kernel(
    const bf16_t* __restrict__ Y2, const bf16_t* __restrict__ Wb,
    const float* __restrict__ biasF, const float* __restrict__ V4,
    const float* __restrict__ c4, float* __restrict__ Out) {
  constexpr int K = 128, KSTEPS = 4;
  __shared__ bf16_t Atile[2][128][40];
  __shared__ bf16_t Btile[2][64][40];
  __shared__ bf16_t Z[128][72];
  __shared__ float sV4[36 * 64];
  __shared__ float sc4[36];

  const int tid  = threadIdx.x;
  const int lane = tid & 31;
  const int wave = tid >> 5;
  const int half = lane >> 4;
  const int m0   = blockIdx.x * 128;

  for (int idx = tid; idx < 36 * 64; idx += 256) sV4[idx] = V4[idx];
  if (tid < 36) sc4[tid] = c4[tid];

  auto stage = [&](int buf, int ks) {
    const int kb = ks * 32;
    for (int idx = tid; idx < 128 * 4; idx += 256) {
      const int r = idx >> 2, c = idx & 3;
      copy16_g2l(Y2 + (size_t)(m0 + r) * K + kb + c * 8, &Atile[buf][r][c * 8]);
    }
    for (int idx = tid; idx < 64 * 4; idx += 256) {
      const int n = idx >> 2, c = idx & 3;
      copy16_g2l(Wb + (size_t)n * K + kb + c * 8, &Btile[buf][n][c * 8]);
    }
  };

  v8f acc[4];
#pragma unroll
  for (int t = 0; t < 4; ++t) acc[t] = zero8();

  stage(0, 0);
  for (int ks = 0; ks < KSTEPS; ++ks) {
    const int cur = ks & 1;
    wait_async_all();
    __syncthreads();
    if (ks + 1 < KSTEPS) stage(cur ^ 1, ks + 1);
    const v16bf a = load_frag(&Atile[cur][wave * 16][0], lane & 15, 40, half);
#pragma unroll
    for (int t = 0; t < 4; ++t) {
      const v16bf b = load_frag(&Btile[cur][t * 16][0], lane & 15, 40, half);
      acc[t] = __builtin_amdgcn_wmma_f32_16x16x32_bf16(false, a, false, b,
                                                       (short)0, acc[t], false, false);
    }
  }
#pragma unroll
  for (int t = 0; t < 4; ++t) {
    const int col = t * 16 + (lane & 15);
    const float bv = biasF[col];
#pragma unroll
    for (int r = 0; r < 8; ++r) {
      float v = acc[t][r] + bv;
      v = v > 0.0f ? v : 0.0f;
      Z[wave * 16 + half * 8 + r][col] = (bf16_t)v;
    }
  }
  __syncthreads();
  for (int o = tid; o < 128 * 36; o += 256) {  // V4: 64 -> 36 per row
    const int m = o / 36, j = o - m * 36;
    float acc2 = sc4[j];
#pragma unroll
    for (int k = 0; k < 64; ++k) acc2 += (float)Z[m][k] * sV4[j * 64 + k];
    Out[(size_t)(m0 + m) * 36 + j] = acc2;
  }
}

__global__ void zero_kernel(float* p, int n) {
  const int i = blockIdx.x * blockDim.x + threadIdx.x;
  if (i < n) p[i] = 0.0f;
}

__global__ void bn_finalize_kernel(const float* __restrict__ sum,
                                   const float* __restrict__ sq,
                                   const float* __restrict__ g,
                                   const float* __restrict__ be, float* a,
                                   float* s, int n, float invB) {
  const int i = blockIdx.x * blockDim.x + threadIdx.x;
  if (i < n) {
    const float m = sum[i] * invB;
    const float v = sq[i] * invB - m * m;
    const float inv = rsqrtf(v + 1e-5f);
    const float ai = g[i] * inv;
    a[i] = ai;
    s[i] = be[i] - m * ai;
  }
}

extern "C" void kernel_launch(void* const* d_in, const int* in_sizes, int n_in,
                              void* d_out, int out_size, void* d_ws,
                              size_t ws_size, hipStream_t stream) {
  constexpr int B = 131072;
  const float* state = (const float*)d_in[0];
  const float* W1 = (const float*)d_in[1];
  const float* b1 = (const float*)d_in[2];
  const float* g1 = (const float*)d_in[3];
  const float* be1 = (const float*)d_in[4];
  const float* W2 = (const float*)d_in[5];
  const float* b2 = (const float*)d_in[6];
  const float* g2 = (const float*)d_in[7];
  const float* be2 = (const float*)d_in[8];
  const float* W3 = (const float*)d_in[9];
  const float* b3 = (const float*)d_in[10];
  const float* W4 = (const float*)d_in[11];
  const float* b4 = (const float*)d_in[12];
  const float* qp = (const float*)d_in[13];
  const float* V1 = (const float*)d_in[14];
  const float* c1 = (const float*)d_in[15];
  const float* h1 = (const float*)d_in[16];
  const float* d1 = (const float*)d_in[17];
  const float* V2 = (const float*)d_in[18];
  const float* c2 = (const float*)d_in[19];
  const float* h2 = (const float*)d_in[20];
  const float* d2 = (const float*)d_in[21];
  const float* V3 = (const float*)d_in[22];
  const float* c3 = (const float*)d_in[23];
  const float* V4 = (const float*)d_in[24];
  const float* c4 = (const float*)d_in[25];

  char* ws = (char*)d_ws;
  float* stats = (float*)ws;                 // 1024 f stats + 1024 f affines + 256 f folded bias
  float* sum1 = stats + 0;   float* sq1 = stats + 128;
  float* sum2 = stats + 256; float* sq2 = stats + 384;
  float* sum3 = stats + 512; float* sq3 = stats + 640;
  float* sum4 = stats + 768; float* sq4 = stats + 896;
  float* aff = stats + 1024;
  float* a1 = aff + 0;   float* s1 = aff + 128;
  float* a2 = aff + 256; float* s2 = aff + 384;
  float* a3 = aff + 512; float* s3 = aff + 640;
  float* a4 = aff + 768; float* s4 = aff + 896;
  float* fb2  = stats + 2048;   // [64]
  float* fbV2 = stats + 2112;   // [128]
  float* fbV3 = stats + 2240;   // [64]
  bf16_t* A1 = (bf16_t*)(ws + 16384);        // [B,128] bf16
  bf16_t* A2 = A1 + (size_t)B * 128;         // [B,64]
  bf16_t* Y1 = A2 + (size_t)B * 64;          // [B,64]
  bf16_t* Y2 = Y1 + (size_t)B * 64;          // [B,128]
  bf16_t* Wb2 = Y2 + (size_t)B * 128;        // [64,128]
  bf16_t* Vb2 = Wb2 + 64 * 128;              // [128,64]
  bf16_t* Vb3 = Vb2 + 128 * 64;              // [64,128]
  float* out = (float*)d_out;
  const float invB = 1.0f / (float)B;

  zero_kernel<<<4, 256, 0, stream>>>(stats, 1024);

  gemm1_kernel<<<B / 128, 256, 0, stream>>>(state, W1, b1, A1, sum1, sq1);
  bn_finalize_kernel<<<1, 128, 0, stream>>>(sum1, sq1, g1, be1, a1, s1, 128, invB);

  fold_weight_kernel<64, 128><<<1, 64, 0, stream>>>(W2, b2, a1, s1, Wb2, fb2);
  gemm_async_kernel<64, 4><<<B / 128, 256, 0, stream>>>(A1, Wb2, fb2, A2, sum2, sq2);
  bn_finalize_kernel<<<1, 64, 0, stream>>>(sum2, sq2, g2, be2, a2, s2, 64, invB);

  middle_kernel<<<B / 256, 256, 0, stream>>>(A2, a2, s2, W3, b3, W4, b4, qp, V1,
                                             c1, Y1, sum3, sq3);
  bn_finalize_kernel<<<1, 64, 0, stream>>>(sum3, sq3, h1, d1, a3, s3, 64, invB);

  fold_weight_kernel<128, 64><<<1, 128, 0, stream>>>(V2, c2, a3, s3, Vb2, fbV2);
  gemm_async_kernel<128, 2><<<B / 128, 256, 0, stream>>>(Y1, Vb2, fbV2, Y2, sum4, sq4);
  bn_finalize_kernel<<<1, 128, 0, stream>>>(sum4, sq4, h2, d2, a4, s4, 128, invB);

  fold_weight_kernel<64, 128><<<1, 64, 0, stream>>>(V3, c3, a4, s4, Vb3, fbV3);
  final_kernel<<<B / 128, 256, 0, stream>>>(Y2, Vb3, fbV3, V4, c4, out);
}
